// AnonTokyoEncoder_18545668784683
// MI455X (gfx1250) — matrix operand order repaired and checked by reading, not compiled, not run on hardware
//
#include <hip/hip_runtime.h>
#include <stdint.h>

// ---------------- problem constants ----------------
#define Bc 8
#define Ac 96
#define Tc 11
#define Mc 512
#define Pc 20
#define Dc 256
#define Hc 8
#define Kc 32

#define RA (Bc*Ac*Tc)    // 8448 agent tokens
#define RM (Bc*Mc)       // 4096 map tokens
#define RMP (Bc*Mc*Pc)   // 81920 map points

typedef __attribute__((ext_vector_type(16))) _Float16 v16h;
typedef __attribute__((ext_vector_type(8)))  _Float16 v8h;
typedef __attribute__((ext_vector_type(8)))  float    v8f;

// ---------------- wave32 helpers ----------------
__device__ __forceinline__ float wred_sum(float v){
  #pragma unroll
  for (int m = 16; m > 0; m >>= 1) v += __shfl_xor(v, m, 32);
  return v;
}
__device__ __forceinline__ float wred_max(float v){
  #pragma unroll
  for (int m = 16; m > 0; m >>= 1) v = fmaxf(v, __shfl_xor(v, m, 32));
  return v;
}

// RoPE on one 32-dim head slice: pairs (2j,2j+1), j<8 -> pos.x, j>=8 -> pos.y
__device__ __forceinline__ void rope32(float* x, float px, float py, float hd){
  const float F[8] = {1.f, 0.31622776601683794f, 0.1f, 0.031622776601683791f,
                      0.01f, 0.0031622776601683794f, 0.001f, 0.00031622776601683794f};
  #pragma unroll
  for (int j = 0; j < 16; ++j){
    float ang = ((j < 8) ? px : py) * F[j & 7] + hd;
    float c = cosf(ang), s = sinf(ang);
    float x1 = x[2*j], x2 = x[2*j+1];
    x[2*j]   = x1*c - x2*s;
    x[2*j+1] = x1*s + x2*c;
  }
}

// ---------------- GEMM prep: fp32 -> fp16 with K padded to multiple of 32 ---------------
// activations: y[r][k] = (k < K) ? x[r*K+k] : 0          (y stride Kp)
__global__ void k_cvt16(const float* __restrict__ x, _Float16* __restrict__ y,
                        int rows, int K, int Kp)
{
  size_t e = (size_t)blockIdx.x * blockDim.x + threadIdx.x;
  if (e >= (size_t)rows * Kp) return;
  int k = (int)(e % Kp);
  size_t r = e / Kp;
  y[e] = (k < K) ? (_Float16)x[r * K + k] : (_Float16)0.f;
}
// weights: WT[n][k] = (k < K) ? W[k*N+n] : 0             (WT stride Kp) -> contiguous K runs
__global__ void k_wt16(const float* __restrict__ W, _Float16* __restrict__ WT,
                       int K, int N, int Kp)
{
  size_t e = (size_t)blockIdx.x * blockDim.x + threadIdx.x;
  if (e >= (size_t)N * Kp) return;
  int k = (int)(e % Kp);
  int n = (int)(e / Kp);
  WT[e] = (k < K) ? (_Float16)W[(size_t)k * N + n] : (_Float16)0.f;
}

// ---------------- WMMA GEMM: C[M,N] = A16[M,Kp] @ WT16[N,Kp]^T + bias ------------------
// One wave computes a 16(M) x 64(N) tile: 4 v_wmma per K-step sharing one A fragment.
// Fragment layouts used here (wave32; A and B share contiguous k = 16g + s):
//   A 16x32 f16: lane(m=lane&15, g=lane>>4): halves 0..7 = A[m][k0+16g .. +8),
//                halves 8..15 = A[m][k0+16g+8 .. +8)        -> two 16B loads
//   B 32x16 f16: lane(n, g): halves 0..15 = Bt[n][k0+16g .. +16) -> one 32B load
//   C/D 16x16 f32: VGPR r -> row r+8g, col lane&15
__global__ void k_gemm_wmma(const _Float16* __restrict__ A16, const _Float16* __restrict__ WT,
                            const float* __restrict__ bias, float* __restrict__ Cout,
                            int Mrows, int Kp, int Ndim, int relu,
                            const unsigned char* __restrict__ rowmask)
{
  int wave = (blockIdx.x * blockDim.x + threadIdx.x) >> 5;
  int lane = threadIdx.x & 31;
  int ntn  = Ndim >> 6;                       // 64-wide N tiles (all N here are mult. of 64)
  int tm = wave / ntn;
  int tn = wave - tm * ntn;
  if (tm * 16 >= Mrows) return;               // wave-uniform guard (EXEC stays full)
  int ml = lane & 15;
  int g  = lane >> 4;

  const _Float16* ar  = A16 + (size_t)(tm * 16 + ml) * Kp;
  const _Float16* wt0 = WT  + (size_t)(tn * 64 + ml) * Kp;

  v8f z = {0.f,0.f,0.f,0.f,0.f,0.f,0.f,0.f};
  v8f acc0 = z, acc1 = z, acc2 = z, acc3 = z;

  for (int k0 = 0; k0 < Kp; k0 += 32) {
    __builtin_prefetch(ar + k0 + 32, 0, 1);   // global_prefetch_b8
    v8h alo = *(const v8h*)(ar + k0 + 16 * g);
    v8h ahi = *(const v8h*)(ar + k0 + 16 * g + 8);
    v16h af = __builtin_shufflevector(alo, ahi,
                                      0,1,2,3,4,5,6,7,8,9,10,11,12,13,14,15);
    v16h b0 = *(const v16h*)(wt0 +  0 * (size_t)Kp + k0 + 16 * g);
    v16h b1 = *(const v16h*)(wt0 + 16 * (size_t)Kp + k0 + 16 * g);
    v16h b2 = *(const v16h*)(wt0 + 32 * (size_t)Kp + k0 + 16 * g);
    v16h b3 = *(const v16h*)(wt0 + 48 * (size_t)Kp + k0 + 16 * g);
    acc0 = __builtin_amdgcn_wmma_f32_16x16x32_f16(false, af, false, b0, (short)0, acc0, false, false);
    acc1 = __builtin_amdgcn_wmma_f32_16x16x32_f16(false, af, false, b1, (short)0, acc1, false, false);
    acc2 = __builtin_amdgcn_wmma_f32_16x16x32_f16(false, af, false, b2, (short)0, acc2, false, false);
    acc3 = __builtin_amdgcn_wmma_f32_16x16x32_f16(false, af, false, b3, (short)0, acc3, false, false);
  }

  v8f accs[4] = {acc0, acc1, acc2, acc3};
  #pragma unroll
  for (int r = 0; r < 8; ++r) {
    int row = tm * 16 + r + 8 * g;
    float rmv = rowmask ? (float)rowmask[row] : 1.f;
    #pragma unroll
    for (int s = 0; s < 4; ++s) {
      int ncol = tn * 64 + s * 16 + ml;
      float v = accs[s][r] + (bias ? bias[ncol] : 0.f);
      if (relu) v = fmaxf(v, 0.f);
      v *= rmv;
      Cout[(size_t)row * Ndim + ncol] = v;
    }
  }
}

// ---------------- LayerNorm over D=256 with residual: out = LN(x + res) ----------------
__global__ void k_ln(const float* __restrict__ x, const float* __restrict__ res,
                     const float* __restrict__ gamma, const float* __restrict__ beta,
                     float* __restrict__ out, int rows)
{
  int row = blockIdx.x * (blockDim.x >> 5) + (threadIdx.x >> 5);
  if (row >= rows) return;
  int lane = threadIdx.x & 31;
  float v[8];
  float s = 0.f;
  #pragma unroll
  for (int i = 0; i < 8; ++i) {
    int c = i * 32 + lane;
    float t = x[(size_t)row * Dc + c];
    if (res) t += res[(size_t)row * Dc + c];
    v[i] = t; s += t;
  }
  s = wred_sum(s);
  float mean = s * (1.f / 256.f);
  float vs = 0.f;
  #pragma unroll
  for (int i = 0; i < 8; ++i) { float d = v[i] - mean; vs += d * d; }
  vs = wred_sum(vs);
  float rstd = rsqrtf(vs * (1.f / 256.f) + 1e-5f);
  #pragma unroll
  for (int i = 0; i < 8; ++i) {
    int c = i * 32 + lane;
    out[(size_t)row * Dc + c] = (v[i] - mean) * rstd * gamma[c] + beta[c];
  }
}

// ---------------- top-K=32 nearest (matches jax.lax.top_k(-d) tie-breaking) -------------
__global__ void k_topk(const float* __restrict__ qpos, const float* __restrict__ kpos,
                       const unsigned char* __restrict__ kmask, int* __restrict__ idx,
                       int nbatch, int nq, int nk, int bdiv)
{
  int t = blockIdx.x * blockDim.x + threadIdx.x;
  if (t >= nbatch * nq) return;
  int kb = (t / nq) / bdiv;
  float qx = qpos[(size_t)t * 2], qy = qpos[(size_t)t * 2 + 1];
  float bd[Kc]; int bi[Kc];
  for (int i = 0; i < Kc; ++i) { bd[i] = 3.0e38f; bi[i] = i; }
  for (int j = 0; j < nk; ++j) {
    size_t kr = (size_t)kb * nk + j;
    float dx = qx - kpos[kr * 2], dy = qy - kpos[kr * 2 + 1];
    float d = kmask[kr] ? (dx * dx + dy * dy) : 3.0e38f;
    if (d < bd[Kc - 1]) {
      int p = Kc - 1;
      while (p > 0 && d < bd[p - 1]) { bd[p] = bd[p - 1]; bi[p] = bi[p - 1]; --p; }
      bd[p] = d; bi[p] = j;
    }
  }
  for (int i = 0; i < Kc; ++i) idx[(size_t)t * Kc + i] = bi[i];
}

// ---------------- sparse attention (one wave per (query,head)) --------------------------
// key row = ((qrow / nq_per_batch) / bdiv) * nk + idx[qrow][lane]
__global__ void k_sparse_attn(const float* __restrict__ q, const float* __restrict__ kbuf,
                              const float* __restrict__ vbuf,
                              const float* __restrict__ qpos, const float* __restrict__ qhd,
                              const float* __restrict__ kpos, const float* __restrict__ khd,
                              const unsigned char* __restrict__ kmask,
                              const int* __restrict__ idx, float* __restrict__ out,
                              int rows_q, int nq_per_batch, int nk, int bdiv)
{
  int wave = (blockIdx.x * blockDim.x + threadIdx.x) >> 5;
  int lane = threadIdx.x & 31;
  int qrow = wave >> 3;
  int h = wave & 7;
  if (qrow >= rows_q) return;
  int kb = (qrow / nq_per_batch) / bdiv;
  int kid = idx[(size_t)qrow * Kc + lane];
  int krow = kb * nk + kid;

  float qv[32], kv[32];
  #pragma unroll
  for (int d = 0; d < 32; ++d) qv[d] = q[(size_t)qrow * Dc + h * 32 + d];
  rope32(qv, qpos[(size_t)qrow * 2], qpos[(size_t)qrow * 2 + 1], qhd[qrow]);
  #pragma unroll
  for (int d = 0; d < 32; ++d) kv[d] = kbuf[(size_t)krow * Dc + h * 32 + d];
  rope32(kv, kpos[(size_t)krow * 2], kpos[(size_t)krow * 2 + 1], khd[krow]);

  float sc = 0.f;
  #pragma unroll
  for (int d = 0; d < 32; ++d) sc += qv[d] * kv[d];
  sc *= 0.17677669529663687f;                 // 1/sqrt(32)
  if (!kmask[krow]) sc = -1e9f;

  float mx = wred_max(sc);
  float e  = expf(sc - mx);
  float ss = wred_sum(e);
  float a  = e / ss;

  float o = 0.f;
  #pragma unroll
  for (int t = 0; t < 32; ++t) {
    float at = __shfl(a, t, 32);
    int   kt = __shfl(krow, t, 32);
    o += at * vbuf[(size_t)kt * Dc + h * 32 + lane];
  }
  out[(size_t)qrow * Dc + h * 32 + lane] = o;
}

// ---------------- temporal causal attention (one wave per (seq,head,qt)) ----------------
// qkv rows laid out (n*T + t) x 768 : [q | k | v]
__global__ void k_temporal(const float* __restrict__ qkv, const unsigned char* __restrict__ valid,
                           float* __restrict__ out, int Nseq)
{
  int wave = (blockIdx.x * blockDim.x + threadIdx.x) >> 5;
  int lane = threadIdx.x & 31;
  int qt = wave % Tc;
  int r2 = wave / Tc;
  int h  = r2 & 7;
  int n  = r2 >> 3;
  if (n >= Nseq) return;
  size_t qrow = (size_t)n * Tc + qt;

  float qv[32];
  #pragma unroll
  for (int d = 0; d < 32; ++d) qv[d] = qkv[qrow * 768 + h * 32 + d];

  int kt = lane;
  bool act = kt < Tc;
  float sc = -3.0e38f;
  if (act) {
    size_t krow = (size_t)n * Tc + kt;
    float s = 0.f;
    #pragma unroll
    for (int d = 0; d < 32; ++d) s += qv[d] * qkv[krow * 768 + 256 + h * 32 + d];
    s *= 0.17677669529663687f;
    bool m = (kt <= qt) && valid[krow];
    sc = m ? s : -1e9f;
  }
  float mx = wred_max(sc);
  float e  = act ? expf(sc - mx) : 0.f;
  float ss = wred_sum(e);
  float a  = e / ss;

  float o = 0.f;
  #pragma unroll
  for (int t = 0; t < Tc; ++t) {
    float at = __shfl(a, t, 32);
    o += at * qkv[((size_t)n * Tc + t) * 768 + 512 + h * 32 + lane];
  }
  out[qrow * Dc + h * 32 + lane] = o;
}

// ---------------- feature builders ----------------
__global__ void k_aug_agent(const float* __restrict__ traj, const unsigned char* __restrict__ mask,
                            float* __restrict__ out)
{
  int i = blockIdx.x * blockDim.x + threadIdx.x;
  if (i >= RA) return;
  int t = i % Tc;
  const float* tr = traj + (size_t)i * 10;
  const float* tb = traj + (size_t)(i - t) * 10;   // row (b,a,t=0)
  float m = mask[i] ? 1.f : 0.f;
  float o[26];
  o[0] = tr[2]; o[1] = tr[3]; o[2] = tr[4]; o[3] = tr[5];     // shape
  o[4] = 0.f; o[5] = 0.f; o[6] = 0.f; o[7] = 0.f; o[8] = 0.f; // type one-hot (zeros)
  #pragma unroll
  for (int j = 0; j < Tc; ++j) o[9 + j] = (j == t) ? 1.f : 0.f; // eye(T)
  o[20] = (float)t * 0.1f;                                   // linspace(0,1,11)
  o[21] = tr[8]; o[22] = tr[9];                              // vel
  int ta = (t == 0) ? 1 : t;                                 // acce[0]=acce[1]
  o[23] = (tb[ta * 10 + 8] - tb[(ta - 1) * 10 + 8]) * 10.f;
  o[24] = (tb[ta * 10 + 9] - tb[(ta - 1) * 10 + 9]) * 10.f;
  for (int j = 0; j < 25; ++j) out[(size_t)i * 26 + j] = o[j] * m;
  out[(size_t)i * 26 + 25] = m;                              // mask channel (post-multiply)
}

__global__ void k_aug_map(const float* __restrict__ polys, const float* __restrict__ center,
                          const float* __restrict__ heading, const unsigned char* __restrict__ mask,
                          float* __restrict__ out)
{
  int i = blockIdx.x * blockDim.x + threadIdx.x;
  if (i >= RMP) return;
  int p = i % Pc;
  int bm = i / Pc;
  float c = cosf(heading[bm]), s = sinf(heading[bm]);
  float cx = center[(size_t)bm * 2], cy = center[(size_t)bm * 2 + 1];
  const float* pl = polys + (size_t)i * 7;
  float x = pl[0] - cx, y = pl[1] - cy;
  float lx = x * c + y * s, ly = -x * s + y * c;
  float ldx = pl[3] * c + pl[4] * s, ldy = -pl[3] * s + pl[4] * c;
  int pp = (p == 0) ? 0 : p - 1;                               // pre = prev point (clamped)
  const float* pl2 = polys + (size_t)(bm * Pc + pp) * 7;
  float x2 = pl2[0] - cx, y2 = pl2[1] - cy;
  float px_ = x2 * c + y2 * s, py_ = -x2 * s + y2 * c;
  float m = mask[i] ? 1.f : 0.f;
  float o[9] = {lx, ly, pl[2], ldx, ldy, pl[5], pl[6], px_, py_};
  for (int j = 0; j < 9; ++j) out[(size_t)i * 9 + j] = o[j] * m;
}

// masked max-pool over P points: pooled[bm][c]
__global__ void k_pnpool(const float* __restrict__ x, const unsigned char* __restrict__ mask,
                         float* __restrict__ pooled)
{
  int t = blockIdx.x * blockDim.x + threadIdx.x;
  if (t >= RM * 64) return;
  int bm = t >> 6, c = t & 63;
  float v = -1e9f;
  for (int p = 0; p < Pc; ++p) {
    size_t r = (size_t)bm * Pc + p;
    float xv = mask[r] ? x[r * 64 + c] : -1e9f;
    v = fmaxf(v, xv);
  }
  pooled[(size_t)bm * 64 + c] = v;
}

__global__ void k_pncat(const float* __restrict__ x, const float* __restrict__ pooled,
                        float* __restrict__ out)
{
  size_t t = (size_t)blockIdx.x * blockDim.x + threadIdx.x;
  if (t >= (size_t)RMP * 128) return;
  int c = (int)(t & 127);
  size_t r = t >> 7;
  int bm = (int)(r / Pc);
  out[t] = (c < 64) ? x[r * 64 + c] : pooled[(size_t)bm * 64 + (c - 64)];
}

__global__ void k_anymask(const unsigned char* __restrict__ pmask, unsigned char* __restrict__ out)
{
  int bm = blockIdx.x * blockDim.x + threadIdx.x;
  if (bm >= RM) return;
  unsigned char v = 0;
  for (int p = 0; p < Pc; ++p) v |= pmask[(size_t)bm * Pc + p];
  out[bm] = v ? 1 : 0;
}

// (B,A,T) aux in (B*T, A) layout: positions / headings / mask
__global__ void k_prep_aux(const float* __restrict__ traj, const float* __restrict__ head,
                           const unsigned char* __restrict__ mask,
                           float* __restrict__ apos, float* __restrict__ ahd,
                           unsigned char* __restrict__ am)
{
  int i = blockIdx.x * blockDim.x + threadIdx.x;
  if (i >= RA) return;
  int a = i % Ac;
  int bt = i / Ac;
  int t = bt % Tc;
  int b = bt / Tc;
  size_t src = ((size_t)b * Ac + a) * Tc + t;
  apos[(size_t)i * 2]     = traj[src * 10 + 0];
  apos[(size_t)i * 2 + 1] = traj[src * 10 + 1];
  ahd[i] = head[(size_t)b * Ac + a];
  am[i]  = mask[src];
}

__global__ void k_bat2bta(const float* __restrict__ in, float* __restrict__ out)
{
  size_t e = (size_t)blockIdx.x * blockDim.x + threadIdx.x;
  if (e >= (size_t)RA * Dc) return;
  int d = (int)(e & 255);
  int r = (int)(e >> 8);               // r = (b*A + a)*T + t
  int t = r % Tc; int r2 = r / Tc; int a = r2 % Ac; int b = r2 / Ac;
  size_t ro = ((size_t)b * Tc + t) * Ac + a;
  out[ro * Dc + d] = in[e];
}

__global__ void k_bta2bat(const float* __restrict__ in, float* __restrict__ out)
{
  size_t e = (size_t)blockIdx.x * blockDim.x + threadIdx.x;
  if (e >= (size_t)RA * Dc) return;
  int d = (int)(e & 255);
  int r = (int)(e >> 8);               // r = (b*A + a)*T + t  (output layout)
  int t = r % Tc; int r2 = r / Tc; int a = r2 % Ac; int b = r2 / Ac;
  size_t ri = ((size_t)b * Tc + t) * Ac + a;
  out[e] = in[ri * Dc + d];
}

__global__ void k_maskmul(float* __restrict__ x, const unsigned char* __restrict__ mask, int rows)
{
  size_t e = (size_t)blockIdx.x * blockDim.x + threadIdx.x;
  if (e >= (size_t)rows * Dc) return;
  x[e] *= (float)mask[e >> 8];
}

__global__ void k_copy(const float* __restrict__ src, float* __restrict__ dst, size_t n)
{
  size_t e = (size_t)blockIdx.x * blockDim.x + threadIdx.x;
  if (e < n) dst[e] = src[e];
}

// ---------------- parameter leaf indices (jax pytree: dict keys sorted) -----------------
// top-level sorted: layers, pn_out, pn_post, pn_pre, tok
// layer sorted: aa_attn, aa_ffn, aa_ffn_norm, aa_norm, am_*, mm_*, tmp
// attn sorted: k,o,q,v ; lin sorted: b,w ; ln sorted: b,g
enum {
  AA_KB=0, AA_KW, AA_OB, AA_OW, AA_QB, AA_QW, AA_VB, AA_VW,
  AA_L1B, AA_L1W, AA_L2B, AA_L2W,
  AA_FNB, AA_FNG, AA_NB, AA_NG,
  AM_KB, AM_KW, AM_OB, AM_OW, AM_QB, AM_QW, AM_VB, AM_VW,
  AM_L1B, AM_L1W, AM_L2B, AM_L2W,
  AM_FNB, AM_FNG, AM_NB, AM_NG,
  MM_KB, MM_KW, MM_OB, MM_OW, MM_QB, MM_QW, MM_VB, MM_VW,
  MM_L1B, MM_L1W, MM_L2B, MM_L2W,
  MM_FNB, MM_FNG, MM_NB, MM_NG,
  T_INB, T_INW, T_L1B, T_L1W, T_L2B, T_L2W,
  T_LN1B, T_LN1G, T_LN2B, T_LN2G, T_OUTB, T_OUTW,
  LAYER_LEAVES                               // == 60
};
#define P_BASE   7
#define PN_OUT_B (P_BASE + 2*LAYER_LEAVES)   // 127
#define PN_OUT_W (PN_OUT_B + 1)
#define PP0B (PN_OUT_W + 1)
#define PP0W (PP0B + 1)
#define PP1B (PP0W + 1)
#define PP1W (PP1B + 1)
#define PR0B (PP1W + 1)
#define PR0W (PR0B + 1)
#define PR1B (PR0W + 1)
#define PR1W (PR1B + 1)
#define PR2B (PR1W + 1)
#define PR2W (PR2B + 1)
#define TOK0B (PR2W + 1)
#define TOK0W (TOK0B + 1)
#define TOK1B (TOK0W + 1)
#define TOK1W (TOK1B + 1)

extern "C" void kernel_launch(void* const* d_in, const int* in_sizes, int n_in,
                              void* d_out, int out_size, void* d_ws, size_t ws_size,
                              hipStream_t stream)
{
  (void)in_sizes; (void)n_in; (void)out_size; (void)ws_size;

  const float*         obj_trajs  = (const float*)d_in[0];
  const unsigned char* obj_mask   = (const unsigned char*)d_in[1];
  const float*         obj_head   = (const float*)d_in[2];
  const float*         map_polys  = (const float*)d_in[3];
  const unsigned char* map_mask   = (const unsigned char*)d_in[4];
  const float*         map_center = (const float*)d_in[5];
  const float*         map_head   = (const float*)d_in[6];
  auto pf = [&](int i){ return (const float*)d_in[i]; };

  // ---- workspace bump allocator ----
  uintptr_t wp = (uintptr_t)d_ws;
  auto alloc = [&](size_t bytes)->void* {
    void* p = (void*)wp;
    wp += (bytes + 255) & ~(size_t)255;
    return p;
  };
  float* agent_in  = (float*)alloc((size_t)RA * 26 * 4);
  float* hid       = (float*)alloc((size_t)RA * Dc * 4);   // generic rows x 256 scratch
  float* s1        = (float*)alloc((size_t)RA * Dc * 4);
  float* agent_feat= (float*)alloc((size_t)RA * Dc * 4);
  float* abt       = (float*)alloc((size_t)RA * Dc * 4);
  float* qb        = (float*)alloc((size_t)RA * Dc * 4);
  float* kb        = (float*)alloc((size_t)RA * Dc * 4);
  float* vb        = (float*)alloc((size_t)RA * Dc * 4);
  float* ffnh      = (float*)alloc((size_t)RA * 1024 * 4);
  float* qkv       = (float*)alloc((size_t)RA * 768 * 4);
  float* mapin     = (float*)alloc((size_t)RMP * 9 * 4);
  float* pnA       = (float*)alloc((size_t)RMP * 64 * 4);
  float* pnB       = (float*)alloc((size_t)RMP * 64 * 4);
  float* pncat     = (float*)alloc((size_t)RMP * 128 * 4);
  float* pooled    = (float*)alloc((size_t)RM * 64 * 4);
  float* map_feat  = (float*)alloc((size_t)RM * Dc * 4);
  float* apos      = (float*)alloc((size_t)RA * 2 * 4);
  float* ahd       = (float*)alloc((size_t)RA * 4);
  unsigned char* amaskT = (unsigned char*)alloc((size_t)RA);
  unsigned char* mvalid = (unsigned char*)alloc((size_t)RM);
  int* mm_idx = (int*)alloc((size_t)RM * Kc * 4);
  int* aa_idx = (int*)alloc((size_t)RA * Kc * 4);
  int* am_idx = (int*)alloc((size_t)RA * Kc * 4);
  // fp16 staging for WMMA (shared, stream-ordered): A up to max(RMP*128, RA*1024) elems
  _Float16* a16  = (_Float16*)alloc((size_t)RMP * 128 * 2);
  _Float16* wt16 = (_Float16*)alloc((size_t)1024 * 256 * 2);

  auto elt = [&](size_t n){ return (int)((n + 255) / 256); };

  auto gemm = [&](const float* A, const float* W, const float* bias, float* C,
                  int Mr, int Kd, int Nd, bool relu, const unsigned char* rmask){
    int Kp = (Kd + 31) & ~31;
    k_cvt16<<<elt((size_t)Mr * Kp), 256, 0, stream>>>(A, a16, Mr, Kd, Kp);
    k_wt16<<<elt((size_t)Nd * Kp), 256, 0, stream>>>(W, wt16, Kd, Nd, Kp);
    long tiles = (long)(Mr / 16) * (Nd / 64);
    k_gemm_wmma<<<(int)((tiles + 3) / 4), 128, 0, stream>>>(a16, wt16, bias, C,
                                                            Mr, Kp, Nd, relu ? 1 : 0, rmask);
  };
  auto ln = [&](const float* x, const float* res, const float* g, const float* b,
                float* out, int rows){
    k_ln<<<(rows + 7) / 8, 256, 0, stream>>>(x, res, g, b, out, rows);
  };
  auto sattn = [&](const float* q, const float* k, const float* v,
                   const float* qp, const float* qh, const float* kp, const float* kh,
                   const unsigned char* km, const int* idx, float* out,
                   int rows, int nqb, int nk, int bdiv){
    int blocks = (rows * 8 * 32 + 255) / 256;
    k_sparse_attn<<<blocks, 256, 0, stream>>>(q, k, v, qp, qh, kp, kh, km, idx, out,
                                              rows, nqb, nk, bdiv);
  };

  // ---------------- tokenize agents ----------------
  k_aug_agent<<<elt(RA), 256, 0, stream>>>(obj_trajs, obj_mask, agent_in);
  gemm(agent_in, pf(TOK0W), pf(TOK0B), hid, RA, 26, 256, true, nullptr);
  gemm(hid, pf(TOK1W), pf(TOK1B), agent_feat, RA, 256, 256, false, nullptr);

  // ---------------- map PointNet ----------------
  k_aug_map<<<elt(RMP), 256, 0, stream>>>(map_polys, map_center, map_head, map_mask, mapin);
  gemm(mapin, pf(PR0W), pf(PR0B), pnA, RMP, 9, 64, true, map_mask);
  gemm(pnA,   pf(PR1W), pf(PR1B), pnB, RMP, 64, 64, true, map_mask);
  gemm(pnB,   pf(PR2W), pf(PR2B), pnA, RMP, 64, 64, true, map_mask);
  k_pnpool<<<elt((size_t)RM * 64), 256, 0, stream>>>(pnA, map_mask, pooled);
  k_pncat<<<elt((size_t)RMP * 128), 256, 0, stream>>>(pnA, pooled, pncat);
  gemm(pncat, pf(PP0W), pf(PP0B), pnB, RMP, 128, 64, true, map_mask);
  gemm(pnB,   pf(PP1W), pf(PP1B), pnA, RMP, 64, 64, true, map_mask);
  k_pnpool<<<elt((size_t)RM * 64), 256, 0, stream>>>(pnA, map_mask, pooled);
  k_anymask<<<elt(RM), 256, 0, stream>>>(map_mask, mvalid);
  gemm(pooled, pf(PN_OUT_W), pf(PN_OUT_B), map_feat, RM, 64, 256, false, mvalid);

  // ---------------- aux + top-k neighbor indices ----------------
  k_prep_aux<<<elt(RA), 256, 0, stream>>>(obj_trajs, obj_head, obj_mask, apos, ahd, amaskT);
  k_topk<<<elt(RM), 256, 0, stream>>>(map_center, map_center, mvalid, mm_idx, Bc, Mc, Mc, 1);
  k_topk<<<elt(RA), 256, 0, stream>>>(apos, apos, amaskT, aa_idx, Bc * Tc, Ac, Ac, 1);
  k_topk<<<elt(RA), 256, 0, stream>>>(apos, map_center, mvalid, am_idx, Bc * Tc, Ac, Mc, Tc);

  // ---------------- encoder layers ----------------
  for (int l = 0; l < 2; ++l) {
    int base = P_BASE + l * LAYER_LEAVES;
    auto lp = [&](int off){ return (const float*)d_in[base + off]; };

    // ---- map <- map sparse attention ----
    gemm(map_feat, lp(MM_QW), lp(MM_QB), qb, RM, 256, 256, false, nullptr);
    gemm(map_feat, lp(MM_KW), lp(MM_KB), kb, RM, 256, 256, false, nullptr);
    gemm(map_feat, lp(MM_VW), lp(MM_VB), vb, RM, 256, 256, false, nullptr);
    sattn(qb, kb, vb, map_center, map_head, map_center, map_head, mvalid, mm_idx,
          hid, RM, Mc, Mc, 1);
    gemm(hid, lp(MM_OW), lp(MM_OB), s1, RM, 256, 256, false, nullptr);
    ln(map_feat, s1, lp(MM_NG), lp(MM_NB), map_feat, RM);
    gemm(map_feat, lp(MM_L1W), lp(MM_L1B), ffnh, RM, 256, 1024, true, nullptr);
    gemm(ffnh, lp(MM_L2W), lp(MM_L2B), s1, RM, 1024, 256, false, nullptr);
    ln(map_feat, s1, lp(MM_FNG), lp(MM_FNB), map_feat, RM);

    // ---- agents to (B*T, A) layout ----
    k_bat2bta<<<elt((size_t)RA * Dc), 256, 0, stream>>>(agent_feat, abt);

    // ---- agent <- agent ----
    gemm(abt, lp(AA_QW), lp(AA_QB), qb, RA, 256, 256, false, nullptr);
    gemm(abt, lp(AA_KW), lp(AA_KB), kb, RA, 256, 256, false, nullptr);
    gemm(abt, lp(AA_VW), lp(AA_VB), vb, RA, 256, 256, false, nullptr);
    sattn(qb, kb, vb, apos, ahd, apos, ahd, amaskT, aa_idx, hid, RA, Ac, Ac, 1);
    gemm(hid, lp(AA_OW), lp(AA_OB), s1, RA, 256, 256, false, nullptr);
    ln(abt, s1, lp(AA_NG), lp(AA_NB), abt, RA);
    gemm(abt, lp(AA_L1W), lp(AA_L1B), ffnh, RA, 256, 1024, true, nullptr);
    gemm(ffnh, lp(AA_L2W), lp(AA_L2B), s1, RA, 1024, 256, false, nullptr);
    ln(abt, s1, lp(AA_FNG), lp(AA_FNB), abt, RA);

    // ---- agent <- map (k/v computed once per (b,m); broadcast over T via bdiv) ----
    gemm(abt,      lp(AM_QW), lp(AM_QB), qb, RA, 256, 256, false, nullptr);
    gemm(map_feat, lp(AM_KW), lp(AM_KB), kb, RM, 256, 256, false, nullptr);
    gemm(map_feat, lp(AM_VW), lp(AM_VB), vb, RM, 256, 256, false, nullptr);
    sattn(qb, kb, vb, apos, ahd, map_center, map_head, mvalid, am_idx, hid, RA, Ac, Mc, Tc);
    gemm(hid, lp(AM_OW), lp(AM_OB), s1, RA, 256, 256, false, nullptr);
    ln(abt, s1, lp(AM_NG), lp(AM_NB), abt, RA);
    gemm(abt, lp(AM_L1W), lp(AM_L1B), ffnh, RA, 256, 1024, true, nullptr);
    gemm(ffnh, lp(AM_L2W), lp(AM_L2B), s1, RA, 1024, 256, false, nullptr);
    ln(abt, s1, lp(AM_FNG), lp(AM_FNB), abt, RA);

    // ---- back to (B,A,T) layout ----
    k_bta2bat<<<elt((size_t)RA * Dc), 256, 0, stream>>>(abt, agent_feat);

    // ---- temporal causal attention over T ----
    gemm(agent_feat, lp(T_INW), lp(T_INB), qkv, RA, 256, 768, false, nullptr);
    {
      int waves = Bc * Ac * Hc * Tc;
      k_temporal<<<(waves * 32 + 255) / 256, 256, 0, stream>>>(qkv, obj_mask, hid, Bc * Ac);
    }
    gemm(hid, lp(T_OUTW), lp(T_OUTB), s1, RA, 256, 256, false, nullptr);
    ln(agent_feat, s1, lp(T_LN1G), lp(T_LN1B), agent_feat, RA);
    gemm(agent_feat, lp(T_L1W), lp(T_L1B), ffnh, RA, 256, 1024, true, nullptr);
    gemm(ffnh, lp(T_L2W), lp(T_L2B), s1, RA, 1024, 256, false, nullptr);
    ln(agent_feat, s1, lp(T_LN2G), lp(T_LN2B), agent_feat, RA);
    k_maskmul<<<elt((size_t)RA * Dc), 256, 0, stream>>>(agent_feat, obj_mask, RA);
  }

  // ---------------- outputs: (agent_feat, map_feat) concatenated flat ----------------
  k_copy<<<elt((size_t)RA * Dc), 256, 0, stream>>>(agent_feat, (float*)d_out, (size_t)RA * Dc);
  k_copy<<<elt((size_t)RM * Dc), 256, 0, stream>>>(map_feat, (float*)d_out + (size_t)RA * Dc,
                                                   (size_t)RM * Dc);
}
